// BagSelfAttentionModule_15762529976587
// MI455X (gfx1250) — matrix operand order repaired and checked by reading, not compiled, not run on hardware
//
#include <hip/hip_runtime.h>
#include <stdint.h>

// ---------------------------------------------------------------------------
// BagSelfAttention for MI455X (gfx1250): bf16 WMMA pipeline
// B=8, N=2048, L=1280, H=160
// ---------------------------------------------------------------------------

#define BB   8
#define NN   2048
#define LL   1280
#define HH   160
#define NC   1600   // fused projection cols: [q(160) | k(160) | v(1280)]

typedef __attribute__((ext_vector_type(16))) __bf16 bf16x16;
typedef __attribute__((ext_vector_type(8)))  float  floatx8;

union Frag {
    bf16x16 v;
    uint4   q[2];
};

// float -> bf16 (round to nearest even), stored as raw uint16
__device__ __forceinline__ uint16_t f2bf(float f) {
    uint32_t u = __float_as_uint(f);
    uint32_t r = (u + 0x7FFFu + ((u >> 16) & 1u)) >> 16;
    return (uint16_t)r;
}

// B operand: lane holds column (lane%16); elements 0..7 = K[half*8 ..],
// elements 8..15 = K[16+half*8 ..] (rows strided by ld)
__device__ __forceinline__ bf16x16 load_b_row(const uint16_t* p, size_t ld, int half) {
    Frag f;
    uint16_t u[16];
#pragma unroll
    for (int s = 0; s < 16; ++s) {
        int k = (s < 8) ? (half * 8 + s) : (16 + half * 8 + (s - 8));
        u[s] = p[(size_t)k * ld];
    }
    __builtin_memcpy(&f, u, 32);
    return f.v;
}

// A operand: lane holds row (lane%16); elements 0..7 = K[kbase+half*8 ..],
// elements 8..15 = K[kbase+16+half*8 ..]  (per ISA 16-bit A layout)
__device__ __forceinline__ bf16x16 load_a_frag(const uint16_t* rowp, int kbase, int half) {
    Frag f;
    f.q[0] = *(const uint4*)(rowp + kbase + half * 8);
    f.q[1] = *(const uint4*)(rowp + kbase + 16 + half * 8);
    return f.v;
}

__device__ __forceinline__ floatx8 wmma_bf16(bf16x16 a, bf16x16 b, floatx8 c) {
    // (neg_a, A, neg_b, B, c_mod, C, reuse_a, reuse_b)
    return __builtin_amdgcn_wmma_f32_16x16x32_bf16(false, a, false, b, (short)0, c, false, false);
}

// ---------------------------------------------------------------------------
// Kernel 1: cast x (f32) -> bf16
// ---------------------------------------------------------------------------
__global__ void cast_x_kernel(const float* __restrict__ x, uint16_t* __restrict__ xh, int n) {
    int i = blockIdx.x * blockDim.x + threadIdx.x;
    int stride = gridDim.x * blockDim.x;
    for (; i < n; i += stride) xh[i] = f2bf(x[i]);
}

// ---------------------------------------------------------------------------
// Kernel 2: build fused transposed weight WhT[k=1280][n=1600] (bf16) + bias[1600]
// ---------------------------------------------------------------------------
__global__ void prep_w_kernel(const float* __restrict__ Wq, const float* __restrict__ bq,
                              const float* __restrict__ Wk, const float* __restrict__ bk,
                              const float* __restrict__ Wv, const float* __restrict__ bv,
                              uint16_t* __restrict__ WhT, float* __restrict__ bias) {
    int idx = blockIdx.x * blockDim.x + threadIdx.x;
    int stride = gridDim.x * blockDim.x;
    const int total = LL * NC;
    for (int i = idx; i < total; i += stride) {
        int k = i / NC;
        int n = i - k * NC;
        float wv;
        if (n < HH)            wv = Wq[n * LL + k];
        else if (n < 2 * HH)   wv = Wk[(n - HH) * LL + k];
        else                   wv = Wv[(n - 2 * HH) * LL + k];
        WhT[(size_t)k * NC + n] = f2bf(wv);
    }
    if (idx < NC) {
        float bvv;
        if (idx < HH)          bvv = bq[idx];
        else if (idx < 2 * HH) bvv = bk[idx - HH];
        else                   bvv = bv[idx - 2 * HH];
        bias[idx] = bvv;
    }
}

// ---------------------------------------------------------------------------
// Kernel 3: fused projection GEMM, register-staged double buffering.
//   P[16384][1600] = xh[16384][1280] @ WhT + bias, routed to q / kT / v (bf16)
//   Block: 128 threads (4 waves). Tile: M=64, N=64 (wave -> 16 cols, 4 M-subtiles)
// ---------------------------------------------------------------------------
__global__ __launch_bounds__(128) void proj_gemm_kernel(
        const uint16_t* __restrict__ xh, const uint16_t* __restrict__ WhT,
        const float* __restrict__ bias,
        uint16_t* __restrict__ qh, uint16_t* __restrict__ kT, uint16_t* __restrict__ vh) {
    __shared__ __align__(16) uint16_t a_lds[64 * 32];

    const int tid  = threadIdx.x;
    const int w    = tid >> 5;
    const int l    = tid & 31;
    const int lr   = l & 15;
    const int half = l >> 4;
    const int m0   = blockIdx.y * 64;   // 256 row groups
    const int n0   = blockIdx.x * 64;   // 25 col groups

    const int arow = tid >> 1;          // cooperative A-tile load mapping
    const int ahs  = tid & 1;
    const uint16_t* asrc_base = xh + (size_t)(m0 + arow) * LL + ahs * 16;
    uint16_t* adst = &a_lds[arow * 32 + ahs * 16];

    floatx8 zero = {};
    floatx8 acc[4];
#pragma unroll
    for (int s = 0; s < 4; ++s) acc[s] = zero;

    // prologue: stage k-chunk 0 in registers
    uint4 stg0 = *(const uint4*)(asrc_base);
    uint4 stg1 = *(const uint4*)(asrc_base + 8);

    for (int kc = 0; kc < LL / 32; ++kc) {
        __syncthreads();                    // readers of previous tile done
        *(uint4*)(adst)     = stg0;
        *(uint4*)(adst + 8) = stg1;
        __syncthreads();                    // tile visible to all waves

        // prefetch next k-chunk into registers; overlaps with compute below
        if (kc + 1 < LL / 32) {
            const uint16_t* src = asrc_base + (kc + 1) * 32;
            stg0 = *(const uint4*)(src);
            stg1 = *(const uint4*)(src + 8);
        }

        // B operand: lane l holds col n0+w*16+lr, K rows of chunk kc in table order
        bf16x16 bfr = load_b_row(WhT + (size_t)(kc * 32) * NC + n0 + w * 16 + lr, NC, half);

        // load all 4 A fragments first (ds clause), then WMMA back-to-back
        bf16x16 af[4];
#pragma unroll
        for (int s = 0; s < 4; ++s)
            af[s] = load_a_frag(a_lds + (s * 16 + lr) * 32, 0, half);
#pragma unroll
        for (int s = 0; s < 4; ++s)
            acc[s] = wmma_bf16(af[s], bfr, acc[s]);
    }

    // epilogue: +bias, cast bf16, route to q / kT / v
    const int n  = n0 + w * 16 + lr;
    const float bn = bias[n];
#pragma unroll
    for (int s = 0; s < 4; ++s) {
#pragma unroll
        for (int g = 0; g < 8; ++g) {
            int m = m0 + s * 16 + g + half * 8;       // m = b*2048 + token
            uint16_t hv = f2bf(acc[s][g] + bn);
            if (n < HH) {
                qh[(size_t)m * HH + n] = hv;
            } else if (n < 2 * HH) {
                int hidx = n - HH;
                int bidx = m >> 11;
                int tok  = m & (NN - 1);
                kT[((size_t)bidx * HH + hidx) * NN + tok] = hv;
            } else {
                vh[(size_t)m * LL + (n - 2 * HH)] = hv;
            }
        }
    }
}

// ---------------------------------------------------------------------------
// Kernel 4: attention. One workgroup = (batch b, 16-query tile).
//   Phase 1: S[16][2048] = q kT (WMMA, q frags register-resident) -> LDS (f32)
//   Phase 2: row softmax (wave32 shfl reductions) -> P (bf16, LDS)
//   Phase 3: O[16][1280] = P V (WMMA, batched B loads), epilogue out = O + x
// ---------------------------------------------------------------------------
__global__ __launch_bounds__(256) void attn_kernel(
        const uint16_t* __restrict__ qh, const uint16_t* __restrict__ kT,
        const uint16_t* __restrict__ vh, const float* __restrict__ x,
        float* __restrict__ out) {
    __shared__ __align__(16) uint16_t q_lds[16 * HH];      //   5 KB
    __shared__ __align__(16) float    s_lds[16 * NN];      // 128 KB
    __shared__ __align__(16) uint16_t p_lds[16 * NN];      //  64 KB

    const int qt   = blockIdx.x;           // 0..127
    const int b    = blockIdx.y;           // 0..7
    const int tid  = threadIdx.x;
    const int w    = tid >> 5;
    const int l    = tid & 31;
    const int lr   = l & 15;
    const int half = l >> 4;
    const int i0   = qt * 16;

    // load q tile 16x160 bf16 into LDS
    {
        const uint16_t* src = qh + ((size_t)b * NN + i0) * HH;
        for (int e = tid; e < 16 * HH; e += 256) q_lds[e] = src[e];
    }
    __syncthreads();

    // ---- Phase 1: S = q @ kT ------------------------------------------------
    // hoist the 5 loop-invariant q A-fragments into registers (40 VGPRs)
    bf16x16 qf[HH / 32];
#pragma unroll
    for (int hc = 0; hc < HH / 32; ++hc)
        qf[hc] = load_a_frag(q_lds + lr * HH, hc * 32, half);

    for (int t = 0; t < 16; ++t) {
        const int j0 = (w + 8 * t) * 16;
        floatx8 acc = {};
        bf16x16 kb[HH / 32];
#pragma unroll
        for (int hc = 0; hc < HH / 32; ++hc)
            kb[hc] = load_b_row(kT + ((size_t)b * HH + hc * 32) * NN + j0 + lr, NN, half);
#pragma unroll
        for (int hc = 0; hc < HH / 32; ++hc)
            acc = wmma_bf16(qf[hc], kb[hc], acc);

        const int col = j0 + lr;
#pragma unroll
        for (int g = 0; g < 8; ++g) s_lds[(g + half * 8) * NN + col] = acc[g];
    }
    __syncthreads();

    // ---- Phase 2: softmax (each wave handles rows w and w+8) ---------------
    for (int r = w; r < 16; r += 8) {
        float m = -3.402823466e38f;
        for (int j = l; j < NN; j += 32) m = fmaxf(m, s_lds[r * NN + j]);
#pragma unroll
        for (int off = 16; off > 0; off >>= 1) m = fmaxf(m, __shfl_xor(m, off, 32));
        float sum = 0.f;
        for (int j = l; j < NN; j += 32) {
            float e = __expf(s_lds[r * NN + j] - m);
            s_lds[r * NN + j] = e;
            sum += e;
        }
#pragma unroll
        for (int off = 16; off > 0; off >>= 1) sum += __shfl_xor(sum, off, 32);
        const float inv = 1.f / sum;
        for (int j = l; j < NN; j += 32)
            p_lds[r * NN + j] = f2bf(s_lds[r * NN + j] * inv);
    }
    __syncthreads();

    // ---- Phase 3: O = P @ V (wave w owns cols [w*160, w*160+160)) ----------
    floatx8 zero = {};
    floatx8 acc[10];
#pragma unroll
    for (int t = 0; t < 10; ++t) acc[t] = zero;
    const int c0 = w * 160;

    for (int jc = 0; jc < NN / 32; ++jc) {
        const int k0 = jc * 32;
        bf16x16 afr = load_a_frag(p_lds + lr * NN, k0, half);
        const uint16_t* vrow = vh + ((size_t)b * NN + k0) * LL + c0 + lr;

        // prefetch the V rows 8 k-chunks ahead (stream stays warm in L2)
        if (jc + 8 < NN / 32) __builtin_prefetch(vrow + (size_t)8 * 32 * LL, 0, 0);

        // batch the 10 B-fragment loads (vmem clause), then 10 WMMAs
        bf16x16 vb[10];
#pragma unroll
        for (int t = 0; t < 10; ++t) vb[t] = load_b_row(vrow + t * 16, LL, half);
#pragma unroll
        for (int t = 0; t < 10; ++t) acc[t] = wmma_bf16(afr, vb[t], acc[t]);
    }

    // epilogue: out = O + x   (GAMMA == 1)
#pragma unroll
    for (int t = 0; t < 10; ++t) {
        const int col = c0 + t * 16 + lr;
#pragma unroll
        for (int g = 0; g < 8; ++g) {
            const int row = i0 + g + half * 8;
            const size_t idx = ((size_t)b * NN + row) * LL + col;
            out[idx] = acc[t][g] + x[idx];
        }
    }
}

// ---------------------------------------------------------------------------
// Host-side launch
// ---------------------------------------------------------------------------
extern "C" void kernel_launch(void* const* d_in, const int* in_sizes, int n_in,
                              void* d_out, int out_size, void* d_ws, size_t ws_size,
                              hipStream_t stream) {
    const float* x  = (const float*)d_in[0];
    const float* Wq = (const float*)d_in[1];
    const float* bq = (const float*)d_in[2];
    const float* Wk = (const float*)d_in[3];
    const float* bk = (const float*)d_in[4];
    const float* Wv = (const float*)d_in[5];
    const float* bv = (const float*)d_in[6];
    float* out = (float*)d_out;

    // workspace layout (bytes)
    char* ws = (char*)d_ws;
    uint16_t* xh   = (uint16_t*)(ws);                              // 41,943,040
    uint16_t* WhT  = (uint16_t*)(ws + 41943040);                   //  4,096,000
    float*    bias = (float*)   (ws + 46039040);                   //      6,400
    uint16_t* qh   = (uint16_t*)(ws + 46045440);                   //  5,242,880
    uint16_t* kT   = (uint16_t*)(ws + 51288320);                   //  5,242,880
    uint16_t* vh   = (uint16_t*)(ws + 56531200);                   // 41,943,040

    cast_x_kernel<<<2048, 256, 0, stream>>>(x, xh, BB * NN * LL);
    prep_w_kernel<<<2048, 256, 0, stream>>>(Wq, bq, Wk, bk, Wv, bv, WhT, bias);
    proj_gemm_kernel<<<dim3(NC / 64, (BB * NN) / 64), 128, 0, stream>>>(
        xh, WhT, bias, qh, kT, vh);
    attn_kernel<<<dim3(NN / 16, BB), 256, 0, stream>>>(qh, kT, vh, x, out);
}
